// HiddenMarkovModel_22265110463208
// MI455X (gfx1250) — matrix operand order repaired and checked by reading, not compiled, not run on hardware
//
#include <hip/hip_runtime.h>

typedef __attribute__((ext_vector_type(16))) _Float16 v16h;
typedef __attribute__((ext_vector_type(8)))  float    v8f;

#define KTAGS 128
#define DDIM  256
#define NEGINF (-1e9f)
#define APAD 132   // alpha row stride (floats): bank stride 4
#define PPAD 136   // pbuf  row stride (halfs):  272B rows, bank stride 4
#define LN2F     0.6931471805599453f
#define LOG2EF   1.4426950408889634f

// fast raw transcendentals for the hot loop (v_exp_f32 / v_log_f32)
__device__ __forceinline__ float fast_exp(float x) {
  return __builtin_amdgcn_exp2f(x * LOG2EF);
}
__device__ __forceinline__ float fast_log(float x) {
  return __builtin_amdgcn_logf(x) * LN2F;
}

// ---------------------------------------------------------------------------
// Kernel 1: logA = row log_softmax(WA with col 126 = NEG); Aexp = exp(logA) f16
// ---------------------------------------------------------------------------
__global__ void k_build_logA(const float* __restrict__ WA,
                             float* __restrict__ logA,
                             _Float16* __restrict__ AexpH) {
  const int r = threadIdx.x;
  float mx = -3.4e38f;
  for (int c = 0; c < KTAGS; ++c) {
    float x = (c == 126) ? NEGINF : WA[r * KTAGS + c];
    mx = fmaxf(mx, x);
  }
  float s = 0.f;
  for (int c = 0; c < KTAGS; ++c) {
    float x = (c == 126) ? NEGINF : WA[r * KTAGS + c];
    s += __expf(x - mx);
  }
  const float ls = __logf(s);
  for (int c = 0; c < KTAGS; ++c) {
    float x = (c == 126) ? NEGINF : WA[r * KTAGS + c];
    float la = x - mx - ls;
    logA[r * KTAGS + c] = la;
    AexpH[r * KTAGS + c] = (_Float16)__expf(la);
  }
}

// ---------------------------------------------------------------------------
// Kernel 2: per-tag emission softmax stats over V (max + log-sum-exp).
// grid = 128 (one tag per block), 256 threads = 8 waves; wave-cooperative
// coalesced dot products over E (E stays resident in the 192MB L2).
// ---------------------------------------------------------------------------
__global__ __launch_bounds__(256) void k_emis_stats(
    const float* __restrict__ ThetaB, const float* __restrict__ E,
    int V, float* __restrict__ stats) {
  const int t = blockIdx.x;
  __shared__ float th[DDIM];
  __shared__ float red[8];
  const int tid = threadIdx.x, lane = tid & 31, wid = tid >> 5;
  th[tid] = ThetaB[t * DDIM + tid];
  __syncthreads();

  // pass 1: max of logits
  float mx = -3.4e38f;
  for (int v = wid; v < V; v += 8) {
    float part = 0.f;
    const float* er = E + (size_t)v * DDIM + lane * 8;
    #pragma unroll
    for (int i = 0; i < 8; ++i) part += th[lane * 8 + i] * er[i];
    for (int m = 1; m < 32; m <<= 1) part += __shfl_xor(part, m, 32);
    mx = fmaxf(mx, part);                 // identical across lanes of wave
  }
  if (lane == 0) red[wid] = mx;
  __syncthreads();
  if (tid == 0) {
    float g = red[0];
    for (int i = 1; i < 8; ++i) g = fmaxf(g, red[i]);
    red[0] = g;
  }
  __syncthreads();
  const float gmx = red[0];
  __syncthreads();

  // pass 2: sum of exp
  float s = 0.f;
  for (int v = wid; v < V; v += 8) {
    float part = 0.f;
    const float* er = E + (size_t)v * DDIM + lane * 8;
    #pragma unroll
    for (int i = 0; i < 8; ++i) part += th[lane * 8 + i] * er[i];
    for (int m = 1; m < 32; m <<= 1) part += __shfl_xor(part, m, 32);
    s += __expf(part - gmx);              // identical across lanes
  }
  if (lane == 0) red[wid] = s;
  __syncthreads();
  if (tid == 0) {
    float ts = 0.f;
    for (int i = 0; i < 8; ++i) ts += red[i];
    stats[t * 2 + 0] = gmx;
    stats[t * 2 + 1] = __logf(ts);
  }
}

// ---------------------------------------------------------------------------
// Kernel 3: logBT[v][t] = ThetaB[t]·E[v] - max_t - lse_t  (NEG for BOS/EOS).
// grid = V, 128 threads (one tag each); coalesced contiguous writes.
// ---------------------------------------------------------------------------
__global__ void k_write_logBT(const float* __restrict__ ThetaB,
                              const float* __restrict__ E, int V,
                              const float* __restrict__ stats,
                              float* __restrict__ logBT) {
  __shared__ float ev[DDIM];
  const int v = blockIdx.x, tid = threadIdx.x;
  ev[tid]       = E[(size_t)v * DDIM + tid];
  ev[tid + 128] = E[(size_t)v * DDIM + tid + 128];
  __syncthreads();
  float d = 0.f;
  const float* th = ThetaB + (size_t)tid * DDIM;
  for (int i = 0; i < DDIM; ++i) d += th[i] * ev[i];
  float r = (tid >= 126) ? NEGINF : (d - stats[tid * 2] - stats[tid * 2 + 1]);
  logBT[(size_t)v * KTAGS + tid] = r;
}

// ---------------------------------------------------------------------------
// Kernel 4: forward recursion. One block = 16 batch rows; alpha resident in
// LDS for all N steps. 8 waves; wave `wid` owns output columns wid*16..+15.
// 3 barriers/step:
//   A: reduce pmax partials -> rm (redundant per row, barrier-free), publish
//      rowmax, build P = exp(alpha - rm) f16               | sync
//   B-E: 4x v_wmma_f32_16x16x32_f16, pipelined emission gather,
//        newv = rowmax + log(C) + logB, store alpha        | sync
//   F: 128-thread pmax partial scan of new alpha           | sync
// ---------------------------------------------------------------------------
__global__ __launch_bounds__(256) void k_hmm_forward(
    const int* __restrict__ words, const float* __restrict__ logA,
    const _Float16* __restrict__ AexpH, const float* __restrict__ logBT,
    float* __restrict__ out, int N) {
  __shared__ float    alpha[16][APAD];
  __shared__ _Float16 pbuf[16][PPAD];
  __shared__ float    rowmax[16];
  __shared__ float    pmax[16][8];

  const int tid  = threadIdx.x;
  const int lane = tid & 31;
  const int wid  = tid >> 5;        // wave id == output column tile
  const int hi   = lane >> 4;       // half-wave select
  const int lo   = lane & 15;       // A-frag row M / B,C-frag col N
  const int b0   = blockIdx.x * 16;
  const unsigned col = wid * 16 + lo;   // this lane's output column

  // Hoist transition-matrix B fragments (constant over time).
  // B 32x16 f16 layout: lane holds col N=lo; K = hi*8 + (e<8 ? e : e+8) within a k-tile.
  v16h bfrag[4];
  {
    const int kb = hi * 8;
    for (int kt = 0; kt < 4; ++kt) {
      #pragma unroll
      for (int e = 0; e < 16; ++e)
        bfrag[kt][e] = AexpH[(unsigned)((kt * 32 + kb + e + (e >= 8 ? 8 : 0)) * KTAGS) + col];
    }
  }

  // Per-lane word-row offsets (32-bit) for the pipelined gather.
  unsigned wrow[8];
  #pragma unroll
  for (int v = 0; v < 8; ++v) wrow[v] = (unsigned)(b0 + v + 8 * hi) * (unsigned)N;

  // alpha_1[t] = logA[BOS, t] + logBT[w0][t]
  for (int idx = tid; idx < 16 * KTAGS; idx += 256) {
    int r = idx >> 7, t = idx & 127;
    unsigned w0 = (unsigned)words[(unsigned)(b0 + r) * (unsigned)N];
    alpha[r][t] = logA[126 * KTAGS + t] + logBT[w0 * KTAGS + t];
  }
  __syncthreads();
  if (tid < 128) {                       // initial pmax partials
    int r = tid >> 3, seg = tid & 7;
    float mx = alpha[r][seg * 16];
    for (int i = 1; i < 16; ++i) mx = fmaxf(mx, alpha[r][seg * 16 + i]);
    pmax[r][seg] = mx;
  }
  __syncthreads();

  // Prologue of the gather pipeline: emission values for step j=1.
  float lb[8];
  if (N > 1) {
    #pragma unroll
    for (int v = 0; v < 8; ++v) {
      unsigned w = (unsigned)words[wrow[v] + 1u];
      lb[v] = logBT[w * KTAGS + col];
    }
  }

  for (int j = 1; j < N; ++j) {
    // --- A: reduce pmax -> rm; publish rowmax; P = exp(alpha - rm) f16 ----
    {
      int r = tid >> 4, off = (tid & 15) * 8;
      float rm = pmax[r][0];
      #pragma unroll
      for (int i = 1; i < 8; ++i) rm = fmaxf(rm, pmax[r][i]);
      if ((tid & 15) == 0) rowmax[r] = rm;
      #pragma unroll
      for (int i = 0; i < 8; ++i)
        pbuf[r][off + i] = (_Float16)fast_exp(alpha[r][off + i] - rm);
    }
    __syncthreads();                     // (1) P + rowmax visible

    // --- B: C = P @ Aexp via 4 accumulated WMMAs --------------------------
    // A 16x32 f16 layout: lane(m=lo,hi): e->(jj=e/2,h=e&1),
    // K = kt*32 + (jj<4 ? 2jj : 16+2(jj-4)) + h + hi*8
    v8f c = {};
    #pragma unroll
    for (int kt = 0; kt < 4; ++kt) {
      v16h a;
      #pragma unroll
      for (int e = 0; e < 16; ++e) {
        int jj = e >> 1, h = e & 1;
        int base = (jj < 4) ? (2 * jj) : (16 + 2 * (jj - 4));
        a[e] = pbuf[lo][kt * 32 + base + h + hi * 8];
      }
      c = __builtin_amdgcn_wmma_f32_16x16x32_f16(
          false, a, false, bfrag[kt], (short)0, c, false, false);
    }

    // --- C: issue next step's emission gather (pipelined) -----------------
    float lbn[8];
    {
      const unsigned jn = (unsigned)((j + 1 < N) ? (j + 1) : j);
      #pragma unroll
      for (int v = 0; v < 8; ++v) {
        unsigned w = (unsigned)words[wrow[v] + jn];
        lbn[v] = logBT[w * KTAGS + col];
      }
    }

    // --- D/E: new alpha = rowmax + log(C) + logB; store; rotate -----------
    #pragma unroll
    for (int v = 0; v < 8; ++v) {
      float nv = rowmax[v + 8 * hi] + fast_log(c[v]) + lb[v];
      alpha[v + 8 * hi][col] = nv;
      lb[v] = lbn[v];
    }
    __syncthreads();                     // (2) new alpha visible

    // --- F: pmax partial scan of new alpha --------------------------------
    if (tid < 128) {
      int r = tid >> 3, seg = tid & 7;
      float mx = alpha[r][seg * 16];
      for (int i = 1; i < 16; ++i) mx = fmaxf(mx, alpha[r][seg * 16 + i]);
      pmax[r][seg] = mx;
    }
    __syncthreads();                     // (3) pmax ready for next A
  }

  // logZ = logsumexp_s(alpha[s] + logA[s, EOS])   (full f32 for the tail)
  if (tid < 16) {
    float rm = pmax[tid][0];
    for (int i = 1; i < 8; ++i) rm = fmaxf(rm, pmax[tid][i]);
    float s = 0.f;
    for (int t = 0; t < KTAGS; ++t)
      s += __expf(alpha[tid][t] - rm + logA[t * KTAGS + 127]);
    out[b0 + tid] = rm + __logf(s);
  }
}

// ---------------------------------------------------------------------------
extern "C" void kernel_launch(void* const* d_in, const int* in_sizes, int n_in,
                              void* d_out, int out_size, void* d_ws, size_t ws_size,
                              hipStream_t stream) {
  const float* ThetaB = (const float*)d_in[0];   // (K, D)
  const float* WA     = (const float*)d_in[1];   // (K, K)
  const float* E      = (const float*)d_in[2];   // (V, D)
  const int*   words  = (const int*)d_in[3];     // (B, N)

  const int V = in_sizes[2] / DDIM;
  const int B = out_size;
  const int N = in_sizes[3] / B;

  // workspace carve-up
  char* ws = (char*)d_ws;
  float*    logBT = (float*)ws;                                        // V*K f32
  size_t off = (size_t)V * KTAGS * sizeof(float);
  float*    logA  = (float*)(ws + off);    off += KTAGS * KTAGS * sizeof(float);
  _Float16* AexpH = (_Float16*)(ws + off); off += KTAGS * KTAGS * sizeof(_Float16);
  float*    stats = (float*)(ws + off);    // 2*K f32

  k_build_logA<<<1, KTAGS, 0, stream>>>(WA, logA, AexpH);
  k_emis_stats<<<KTAGS, 256, 0, stream>>>(ThetaB, E, V, stats);
  k_write_logBT<<<V, KTAGS, 0, stream>>>(ThetaB, E, V, stats, logBT);
  k_hmm_forward<<<B / 16, 256, 0, stream>>>(words, logA, AexpH, logBT,
                                            (float*)d_out, N);
}
